// TorchConfusionMatrixMetric_6579889897596
// MI455X (gfx1250) — matrix operand order repaired and checked by reading, not compiled, not run on hardware
//
#include <hip/hip_runtime.h>
#include <hip/hip_bf16.h>
#include <stdint.h>

typedef __attribute__((ext_vector_type(16))) _Float16 v16h;
typedef __attribute__((ext_vector_type(8)))  float    v8f;
typedef __attribute__((ext_vector_type(4)))  float    vf4;   // native vector: OK for nontemporal builtin

#define NUM_CLASSES   8
#define N_PIXELS      8388608
#define NBLK          1024
#define NTHR          256
#define WAVES_PER_BLK (NTHR / 32)
#define TOTAL_WAVES   (NBLK * WAVES_PER_BLK)      // 8192
#define ROWS_PER_WAVE (N_PIXELS / TOTAL_WAVES)    // 1024
#define ITERS         (ROWS_PER_WAVE / 32)        // 32

// argmax over 8 contiguous floats; strict '>' keeps first-max semantics like jnp.argmax
__device__ __forceinline__ int argmax8(const vf4 a, const vf4 b) {
  float bv = a.x; int bi = 0;
  if (a.y > bv) { bv = a.y; bi = 1; }
  if (a.z > bv) { bv = a.z; bi = 2; }
  if (a.w > bv) { bv = a.w; bi = 3; }
  if (b.x > bv) { bv = b.x; bi = 4; }
  if (b.y > bv) { bv = b.y; bi = 5; }
  if (b.z > bv) { bv = b.z; bi = 6; }
  if (b.w > bv) { bv = b.w; bi = 7; }
  return bi;
}

// 8-way select of a uniform (SGPR) mask by divergent 3-bit index -> 7 v_cndmask
__device__ __forceinline__ uint32_t mux8(uint32_t m, const uint32_t s[8]) {
  uint32_t t01 = (m & 1u) ? s[1] : s[0];
  uint32_t t23 = (m & 1u) ? s[3] : s[2];
  uint32_t t45 = (m & 1u) ? s[5] : s[4];
  uint32_t t67 = (m & 1u) ? s[7] : s[6];
  uint32_t lo  = (m & 2u) ? t23 : t01;
  uint32_t hi  = (m & 2u) ? t67 : t45;
  return (m & 4u) ? hi : lo;
}

// Expand 16 one-hot bits (bit e -> f16 element e) into a v16h fragment.
// VGPR v holds element 2v in [15:0] and element 2v+1 in [31:16]; 0x3C00 = 1.0h.
__device__ __forceinline__ v16h onehot_frag(uint32_t bits16) {
  union { v16h h; uint32_t u[8]; } r;
#pragma unroll
  for (int v = 0; v < 8; ++v) {
    uint32_t x = bits16 >> (2 * v);
    r.u[v] = ((x & 1u) ? 0x00003C00u : 0u) | ((x & 2u) ? 0x3C000000u : 0u);
  }
  return r.h;
}

__global__ __launch_bounds__(NTHR)
void cm_wmma_main(const float* __restrict__ y_true,
                  const float* __restrict__ y_pred,
                  float* __restrict__ partials) {
  const int tid  = threadIdx.x;
  const int lane = tid & 31;
  const int wave = tid >> 5;
  const int gw   = blockIdx.x * WAVES_PER_BLK + wave;

  const uint32_t hi      = (lane >> 4) & 1u;  // lane in [16,31]
  const uint32_t mrow    = (uint32_t)(lane & 7);
  const uint32_t zeroRow = (lane >> 3) & 1u;  // M/N index in [8,15] -> all-zero one-hot

  // Per-wave contiguous span; lane l owns row base+l of each 32-row chunk.
  const vf4* __restrict__ T =
      (const vf4*)y_true + ((long)gw * ROWS_PER_WAVE + lane) * 2;
  const vf4* __restrict__ P =
      (const vf4*)y_pred + ((long)gw * ROWS_PER_WAVE + lane) * 2;

  v8f c = {};  // 16x16 f32 accumulator (top-left 8x8 = confusion counts)

#pragma unroll 4
  for (int it = 0; it < ITERS; ++it) {
    // 32 bytes/row: two global_load_b128 per array; wave covers 1KB contiguous.
    // One-pass 512MB stream (>> 192MB L2): mark non-temporal to avoid L2 churn.
    vf4 t0 = __builtin_nontemporal_load(T + 0);
    vf4 t1 = __builtin_nontemporal_load(T + 1);
    vf4 p0 = __builtin_nontemporal_load(P + 0);
    vf4 p1 = __builtin_nontemporal_load(P + 1);
    T += 64;  // 32 rows * 2 vf4 per chunk
    P += 64;

    const int ti = argmax8(t0, t1);
    const int pi = argmax8(p0, p1);

    // Uniform per-class lane masks over the wave's 32 pixels (K dimension)
    uint32_t mt[8], mp[8];
#pragma unroll
    for (int cls = 0; cls < 8; ++cls) {
      mt[cls] = (uint32_t)__ballot(ti == cls);
      mp[cls] = (uint32_t)__ballot(pi == cls);
    }

    // Per-lane row/col mask: maskT[M], maskP[N] with M = N = lane&15 (0 for >=8)
    uint32_t mA = zeroRow ? 0u : mux8(mrow, mt);
    uint32_t mB = zeroRow ? 0u : mux8(mrow, mp);

    // A (16x32 f16) per ISA layout:
    //   lanes 0-15 : V0..3 K=0..7,  V4..7 K=16..23
    //   lanes16-31 : V0..3 K=8..15, V4..7 K=24..31
    uint32_t bitsA = ((mA >> (hi * 8)) & 0xFFu)
                   | (((mA >> (16 + hi * 8)) & 0xFFu) << 8);
    // B (32x16 f16): same K order as A (lanes 0-15: K=0..7,16..23; lanes 16-31: K=8..15,24..31)
    uint32_t bitsB = ((mB >> (hi * 8)) & 0xFFu)
                   | (((mB >> (16 + hi * 8)) & 0xFFu) << 8);

    v16h a = onehot_frag(bitsA);
    v16h b = onehot_frag(bitsB);

    // c += one_hot(t)[16x32] @ one_hot(p)[32x16]  -- 32 pixels per WMMA
    c = __builtin_amdgcn_wmma_f32_16x16x32_f16(
        /*neg_a=*/false, a, /*neg_b=*/false, b,
        /*c_mod=*/(short)0, c, /*reuse_a=*/false, /*reuse_b=*/false);
  }

  // Per-block reduction: each wave's 8x8 block lives in lanes 0..7, VGPRs 0..7
  // (lane = column N, VGPR r = row M). No atomics needed.
  __shared__ float part[WAVES_PER_BLK][64];
  if (lane < 8) {
#pragma unroll
    for (int m = 0; m < 8; ++m) part[wave][m * 8 + lane] = c[m];
  }
  __syncthreads();
  if (tid < 64) {
    float s = 0.f;
#pragma unroll
    for (int w = 0; w < WAVES_PER_BLK; ++w) s += part[w][tid];
    partials[blockIdx.x * 64 + tid] = s;
  }
}

__global__ __launch_bounds__(64)
void cm_reduce(const float* __restrict__ partials, float* __restrict__ out) {
  const int b = threadIdx.x;  // 64 threads, one per bin
  float s = 0.f;
  for (int p = 0; p < NBLK; ++p) s += partials[p * 64 + b];  // coalesced across bins
  out[b] = s;
}

extern "C" void kernel_launch(void* const* d_in, const int* in_sizes, int n_in,
                              void* d_out, int out_size, void* d_ws, size_t ws_size,
                              hipStream_t stream) {
  (void)in_sizes; (void)n_in; (void)out_size; (void)ws_size;
  const float* y_true = (const float*)d_in[0];
  const float* y_pred = (const float*)d_in[1];
  float* partials = (float*)d_ws;          // NBLK * 64 floats = 256 KB
  float* out      = (float*)d_out;         // 8x8 confusion matrix, f32

  cm_wmma_main<<<NBLK, NTHR, 0, stream>>>(y_true, y_pred, partials);
  cm_reduce<<<1, 64, 0, stream>>>(partials, out);
}
